// Net_60773787238786
// MI455X (gfx1250) — matrix operand order repaired and checked by reading, not compiled, not run on hardware
//
#include <hip/hip_runtime.h>
#include <hip/hip_bf16.h>
#include <math.h>

#define K3 125
#define KS 5

typedef _Float16 v16h __attribute__((ext_vector_type(16)));
typedef float    v8f  __attribute__((ext_vector_type(8)));
typedef float    v4f  __attribute__((ext_vector_type(4)));

// ---------- monotonic float <-> uint key (for atomic max of floats) ----------
__device__ __forceinline__ unsigned fkey(float f) {
    unsigned u = __float_as_uint(f);
    return (u & 0x80000000u) ? ~u : (u | 0x80000000u);
}
__device__ __forceinline__ float unkey(unsigned k) {
    unsigned u = (k & 0x80000000u) ? (k & 0x7FFFFFFFu) : ~k;
    return __uint_as_float(u);
}

// ---------------------------- utility kernels -------------------------------
__global__ void k_fill_int(int* __restrict__ p, int val, int n) {
    int t = blockIdx.x * blockDim.x + threadIdx.x;
    if (t < n) p[t] = val;
}

__global__ void k_deg(const int* __restrict__ dst, float* __restrict__ deg, int E) {
    int t = blockIdx.x * blockDim.x + threadIdx.x;
    if (t < E) atomicAdd(&deg[dst[t]], 1.0f);
}

// ---- W prep: transpose + zero-pad + f16 convert: Wt[co][k], co<Copad, k<Kpad
__global__ void k_wprep(const float* __restrict__ W, _Float16* __restrict__ Wt,
                        int Kk, int Kpad, int Co, int Copad) {
    int t = blockIdx.x * blockDim.x + threadIdx.x;
    if (t >= Copad * Kpad) return;
    int co = t / Kpad, k = t % Kpad;
    Wt[t] = (co < Co && k < Kk) ? (_Float16)W[(long)k * Co + co] : (_Float16)0.0f;
}

// -------- spline basis scatter: A[dst-lo, kidx, ci] += w * x[src, ci] --------
// A row stride = Kpad (zero-padded tail).
__global__ void k_scatter(const float* __restrict__ x, const float* __restrict__ pos,
                          const int* __restrict__ src, const int* __restrict__ dst,
                          int E, float inv2r, int Cin, int lo, int NC, int Kpad,
                          float* __restrict__ A) {
    int t = blockIdx.x * blockDim.x + threadIdx.x;
    if (t >= E * 8) return;
    int e = t >> 3, corner = t & 7;
    int d = dst[e];
    if (d < lo || d >= lo + NC) return;
    int s = src[e];
    float w = 1.0f;
    int kidx = 0, mul = 1;
#pragma unroll
    for (int dim = 0; dim < 3; ++dim) {
        float delta = pos[d * 3 + dim] - pos[s * 3 + dim];
        float p = fminf(fmaxf(delta * inv2r + 0.5f, 0.0f), 1.0f);
        float u = p * (float)(KS - 1);
        float k0f = fminf(fmaxf(floorf(u), 0.0f), (float)(KS - 2));
        float frac = u - k0f;
        int off = (corner >> dim) & 1;
        w *= off ? frac : (1.0f - frac);
        kidx += ((int)k0f + off) * mul;
        mul *= KS;
    }
    float* Arow = A + (long)(d - lo) * Kpad + kidx * Cin;
    const float* xs = x + (long)s * Cin;
    for (int ci = 0; ci < Cin; ++ci)
        atomicAdd(&Arow[ci], w * xs[ci]);
}

// --------------- A-fragment load: 16 f32 -> v16h, no branches ---------------
// 16-bit A 16x32 operand, this code's contiguous k order (B matches): half-wave h, element j: K = 16h+j
__device__ __forceinline__ v16h a_frag(const float* __restrict__ p, int h) {
    const v4f* q = (const v4f*)(p + 16 * h);
    v4f f0 = q[0], f1 = q[1];
    const v4f* r = (const v4f*)(p + 16 * h + 8);
    v4f f2 = r[0], f3 = r[1];
    v16h a;
#pragma unroll
    for (int j = 0; j < 4; ++j) {
        a[j]      = (_Float16)f0[j];
        a[4 + j]  = (_Float16)f1[j];
        a[8 + j]  = (_Float16)f2[j];
        a[12 + j] = (_Float16)f3[j];
    }
    return a;
}

// -------------------- WMMA GEMM: C[M,N] = A[M,K] * Wt[N,K]^T ----------------
// One wave computes a 32x32 tile: 4 x v_wmma_f32_16x16x32_f16 per K-step.
// A: f32 row-major, row stride Kpad (zero-padded). Wt: f16, N-major, stride Kpad.
// Grid: (M/32, Npad/32). Nstore guards only the output columns.
__global__ void k_gemm(const float* __restrict__ A, const _Float16* __restrict__ Wt,
                       float* __restrict__ C, int M, int Nstore, int Kpad, int ldc) {
    int tm = blockIdx.x * 32, tn = blockIdx.y * 32;
    int lane = threadIdx.x & 31;
    int m = lane & 15;     // row (A) / col (B,D) within 16x16 sub-tile
    int h = lane >> 4;     // half-wave selector
    v8f acc00 = {}, acc01 = {}, acc10 = {}, acc11 = {};
    const float*    a0p = A  + (long)(tm + m)      * Kpad;
    const float*    a1p = A  + (long)(tm + 16 + m) * Kpad;
    const _Float16* b0p = Wt + (long)(tn + m)      * Kpad;
    const _Float16* b1p = Wt + (long)(tn + 16 + m) * Kpad;
    for (int kb = 0; kb < Kpad; kb += 32) {
        __builtin_prefetch(a0p + kb + 128, 0, 1);   // global_prefetch_b8
        __builtin_prefetch(a1p + kb + 128, 0, 1);
        v16h a0 = a_frag(a0p + kb, h);
        v16h a1 = a_frag(a1p + kb, h);
        // B 32x16 operand, same contiguous k order as A: element j: K = kb + 16h + j  (contiguous in Wt)
        v16h b0 = *(const v16h*)(b0p + kb + 16 * h);
        v16h b1 = *(const v16h*)(b1p + kb + 16 * h);
        acc00 = __builtin_amdgcn_wmma_f32_16x16x32_f16(false, a0, false, b0, (short)0, acc00, false, false);
        acc01 = __builtin_amdgcn_wmma_f32_16x16x32_f16(false, a0, false, b1, (short)0, acc01, false, false);
        acc10 = __builtin_amdgcn_wmma_f32_16x16x32_f16(false, a1, false, b0, (short)0, acc10, false, false);
        acc11 = __builtin_amdgcn_wmma_f32_16x16x32_f16(false, a1, false, b1, (short)0, acc11, false, false);
    }
    int col0 = tn + m, col1 = tn + 16 + m;
#pragma unroll
    for (int j = 0; j < 8; ++j) {
        int r0 = tm + j + 8 * h;        // C/D layout: VGPR j -> M=j / M=j+8
        int r1 = r0 + 16;
        if (col0 < Nstore) {
            C[(long)r0 * ldc + col0] = acc00[j];
            C[(long)r1 * ldc + col0] = acc10[j];
        }
        if (col1 < Nstore) {
            C[(long)r0 * ldc + col1] = acc01[j];
            C[(long)r1 * ldc + col1] = acc11[j];
        }
    }
}

// ---- epilogue: out = elu(conv/deg + x_in @ root + b), in place over conv ----
__global__ void k_finalize(float* __restrict__ xo, const float* __restrict__ xi,
                           const float* __restrict__ root, const float* __restrict__ b,
                           const float* __restrict__ deg, int N, int Cin, int Co) {
    int t = blockIdx.x * blockDim.x + threadIdx.x;
    if (t >= N * Co) return;
    int n = t / Co, co = t % Co;
    float acc = xo[t] / fmaxf(deg[n], 1.0f) + b[co];
    const float* xr = xi + (long)n * Cin;
    for (int ci = 0; ci < Cin; ++ci) acc += xr[ci] * root[ci * Co + co];
    xo[t] = (acc > 0.0f) ? acc : expm1f(acc);
}

// --------------------- attention score (column 0 only) ----------------------
__global__ void k_att(const float* __restrict__ x, const float* __restrict__ Wa,
                      const float* __restrict__ ba, float* __restrict__ att,
                      int N, int Cin) {
    int t = blockIdx.x * blockDim.x + threadIdx.x;
    if (t >= N) return;
    float acc = ba[0];
    const float* xr = x + (long)t * Cin;
    for (int ci = 0; ci < Cin; ++ci) acc += xr[ci] * Wa[ci * 2];
    att[t] = acc;
}

// ------------------------- attention max-pool passes ------------------------
__global__ void k_pool1(const float* __restrict__ att, const float* __restrict__ pos,
                        const int* __restrict__ cl, unsigned* __restrict__ mkey,
                        float* __restrict__ cnt, float* __restrict__ ppos, int N) {
    int t = blockIdx.x * blockDim.x + threadIdx.x;
    if (t >= N) return;
    int c = cl[t];
    atomicMax(&mkey[c], fkey(att[t]));
    atomicAdd(&cnt[c], 1.0f);
    atomicAdd(&ppos[c * 3 + 0], pos[t * 3 + 0]);
    atomicAdd(&ppos[c * 3 + 1], pos[t * 3 + 1]);
    atomicAdd(&ppos[c * 3 + 2], pos[t * 3 + 2]);
}

__global__ void k_pool2(const float* __restrict__ att, const int* __restrict__ cl,
                        const unsigned* __restrict__ mkey, int* __restrict__ sel, int N) {
    int t = blockIdx.x * blockDim.x + threadIdx.x;
    if (t >= N) return;
    int c = cl[t];
    if (att[t] >= unkey(mkey[c])) atomicMin(&sel[c], t);
}

__global__ void k_gather(const float* __restrict__ xi, const int* __restrict__ sel,
                         const float* __restrict__ cnt, const float* __restrict__ ppos,
                         float* __restrict__ xo, float* __restrict__ po,
                         int nc, int Co, int Ntot) {
    int t = blockIdx.x * blockDim.x + threadIdx.x;
    if (t >= nc * Co) return;
    int c = t / Co, ch = t % Co;
    int idx = min(sel[c], Ntot - 1);
    xo[t] = xi[(long)idx * Co + ch];
    if (ch == 0) {
        float cc = fmaxf(cnt[c], 1.0f);
        po[c * 3 + 0] = ppos[c * 3 + 0] / cc;
        po[c * 3 + 1] = ppos[c * 3 + 1] / cc;
        po[c * 3 + 2] = ppos[c * 3 + 2] / cc;
    }
}

// ------------------------- final 2x2x2 voxel max pool -----------------------
__global__ void k_vox1(const float* __restrict__ x, const float* __restrict__ pos,
                       unsigned* __restrict__ mxkey, float* __restrict__ cnt) {
    int t = blockIdx.x * blockDim.x + threadIdx.x;   // 256 nodes * 256 channels
    if (t >= 256 * 256) return;
    int n = t >> 8, ch = t & 255;
    int v[3];
#pragma unroll
    for (int j = 0; j < 3; ++j)
        v[j] = (int)fminf(fmaxf(floorf(pos[n * 3 + j] + 0.5f), 0.0f), 1.0f);
    int c = (n >> 3) * 8 + v[0] * 4 + v[1] * 2 + v[2];
    atomicMax(&mxkey[c * 256 + ch], fkey(x[n * 256 + ch]));
    if (ch == 0) atomicAdd(&cnt[c], 1.0f);
}

__global__ void k_vox2(const unsigned* __restrict__ mxkey, const float* __restrict__ cnt,
                       float* __restrict__ mx) {
    int t = blockIdx.x * blockDim.x + threadIdx.x;
    if (t >= 256 * 256) return;
    int c = t >> 8;
    mx[t] = (cnt[c] > 0.5f) ? unkey(mxkey[t]) : 0.0f;
}

// ---------------------------- bias + log_softmax ----------------------------
__global__ void k_lsm(const float* __restrict__ logits, const float* __restrict__ bfc,
                      float* __restrict__ out) {
    int r = threadIdx.x;
    if (r >= 32) return;
    float v[10], mx = -1e30f;
#pragma unroll
    for (int o = 0; o < 10; ++o) { v[o] = logits[r * 10 + o] + bfc[o]; mx = fmaxf(mx, v[o]); }
    float s = 0.0f;
#pragma unroll
    for (int o = 0; o < 10; ++o) s += expf(v[o] - mx);
    float ls = logf(s);
#pragma unroll
    for (int o = 0; o < 10; ++o) out[r * 10 + o] = v[o] - mx - ls;
}

// ------------------------------- host driver --------------------------------
static inline int cdiv(int a, int b) { return (a + b - 1) / b; }

extern "C" void kernel_launch(void* const* d_in, const int* in_sizes, int n_in,
                              void* d_out, int out_size, void* d_ws, size_t ws_size,
                              hipStream_t stream) {
    (void)in_sizes; (void)n_in; (void)out_size; (void)ws_size;
    // ---- workspace layout (floats) ----
    float* ws = (float*)d_ws;
    float*    xA     = ws;                       // 2,097,152
    float*    xB     = xA   + 2097152;           // 2,097,152
    float*    posA   = xB   + 2097152;           // 196,608
    float*    posB   = posA + 196608;            // 196,608
    float*    deg    = posB + 196608;            // 65,536
    float*    att    = deg  + 65536;             // 65,536
    unsigned* mkey   = (unsigned*)(att + 65536); // 65,536
    int*      sel    = (int*)(mkey + 65536);     // 65,536
    float*    cnt    = (float*)(sel + 65536);    // 65,536
    float*    ppos   = cnt  + 65536;             // 196,608
    unsigned* mxkey  = (unsigned*)(ppos + 196608); // 65,536
    float*    logits = (float*)(mxkey + 65536);  // 512
    _Float16* Wt     = (_Float16*)(logits + 512); // 4,194,304 halfs (2,097,152 f32)
    float*    Abuf   = (float*)(Wt + 4194304);   // 12,288,000 (49 MB, level-4 A)

    struct LP { int N, E, Cin, Co, NC; float r; };
    const LP lp[5] = {
        {65536, 1048576,   1,  32, 65536, 0.100f},
        {16384,  262144,  32,  64,  2048, 0.100f},
        { 4096,   65536,  64,  96,  1024, 0.125f},
        { 1024,   16384,  96, 128,  1024, 0.250f},
        {  256,    4096, 128, 256,   256, 0.500f},
    };

    const float* x_cur   = (const float*)d_in[0];
    const float* pos_cur = (const float*)d_in[1];

    for (int i = 0; i < 5; ++i) {
        const LP& L = lp[i];
        const float* W    = (const float*)d_in[2 + 3 * i];
        const float* root = (const float*)d_in[3 + 3 * i];
        const float* bias = (const float*)d_in[4 + 3 * i];
        const int*   ei   = (const int*)d_in[27 + i];
        const int*   srcp = ei;
        const int*   dstp = ei + L.E;
        const int    Kk   = K3 * L.Cin;
        const int    Kpad = ((Kk + 31) / 32) * 32;
        const float  inv2r = 1.0f / (2.0f * L.r);

        hipMemsetAsync(deg, 0, (size_t)L.N * 4, stream);
        k_deg<<<cdiv(L.E, 256), 256, 0, stream>>>(dstp, deg, L.E);
        k_wprep<<<cdiv(L.Co * Kpad, 256), 256, 0, stream>>>(W, Wt, Kk, Kpad, L.Co, L.Co);

        for (int lo = 0; lo < L.N; lo += L.NC) {
            hipMemsetAsync(Abuf, 0, (size_t)L.NC * Kpad * 4, stream);
            k_scatter<<<cdiv(L.E * 8, 256), 256, 0, stream>>>(
                x_cur, pos_cur, srcp, dstp, L.E, inv2r, L.Cin, lo, L.NC, Kpad, Abuf);
            k_gemm<<<dim3(L.NC / 32, L.Co / 32), 32, 0, stream>>>(
                Abuf, Wt, xA + (long)lo * L.Co, L.NC, L.Co, Kpad, L.Co);
        }
        k_finalize<<<cdiv(L.N * L.Co, 256), 256, 0, stream>>>(
            xA, x_cur, root, bias, deg, L.N, L.Cin, L.Co);

        if (i < 4) {
            const float* Wa = (const float*)d_in[17 + 2 * i];
            const float* ba = (const float*)d_in[18 + 2 * i];
            const int*   cl = (const int*)d_in[32 + i];
            const int    nc = lp[i + 1].N;
            float* pos_out = (i % 2 == 0) ? posA : posB;

            k_att<<<cdiv(L.N, 256), 256, 0, stream>>>(xA, Wa, ba, att, L.N, L.Co);
            hipMemsetAsync(mkey, 0, (size_t)nc * 4, stream);     // key(-inf) identity
            hipMemsetAsync(cnt,  0, (size_t)nc * 4, stream);
            hipMemsetAsync(ppos, 0, (size_t)nc * 12, stream);
            k_fill_int<<<cdiv(nc, 256), 256, 0, stream>>>(sel, L.N, nc);
            k_pool1<<<cdiv(L.N, 256), 256, 0, stream>>>(att, pos_cur, cl, mkey, cnt, ppos, L.N);
            k_pool2<<<cdiv(L.N, 256), 256, 0, stream>>>(att, cl, mkey, sel, L.N);
            k_gather<<<cdiv(nc * L.Co, 256), 256, 0, stream>>>(
                xA, sel, cnt, ppos, xB, pos_out, nc, L.Co, L.N);
            x_cur = xB;
            pos_cur = pos_out;
        }
    }

    // ---- voxel max pool (256 nodes, 256 channels, 256 voxels) ----
    hipMemsetAsync(mxkey, 0, 256 * 256 * 4, stream);
    hipMemsetAsync(cnt, 0, 256 * 4, stream);
    k_vox1<<<cdiv(256 * 256, 256), 256, 0, stream>>>(xA, pos_cur, mxkey, cnt);
    k_vox2<<<cdiv(256 * 256, 256), 256, 0, stream>>>(mxkey, cnt, xB); // xB = [32,2048]

    // ---- FC (32x2048 @ 2048x10, padded to 32 cols) with WMMA ----
    const float* Wfc = (const float*)d_in[25];
    const float* bfc = (const float*)d_in[26];
    k_wprep<<<cdiv(32 * 2048, 256), 256, 0, stream>>>(Wfc, Wt, 2048, 2048, 10, 32);
    k_gemm<<<dim3(1, 1), 32, 0, stream>>>(xB, Wt, logits, 32, 10, 2048, 10);
    k_lsm<<<1, 32, 0, stream>>>(logits, bfc, (float*)d_out);
}
